// CrossAttention_32023276159246
// MI455X (gfx1250) — compile-verified, matrix operand order checked by reading, not run on hardware
//
#include <hip/hip_runtime.h>
#include <hip/hip_bf16.h>

// ---------------------------------------------------------------------------
// Types for CDNA5 WMMA
// ---------------------------------------------------------------------------
typedef __attribute__((ext_vector_type(16))) __bf16 v16bf;
typedef __attribute__((ext_vector_type(8)))  float  v8f;
typedef __attribute__((ext_vector_type(4)))  int    v4i;

#define BB 4
#define CC 256
#define NN 4096   // H*W
#define DD 32     // C/8

#if defined(__has_builtin)
#if __has_builtin(__builtin_amdgcn_global_load_async_to_lds_b128)
#define HAVE_ASYNC_LDS 1
#endif
#if __has_builtin(__builtin_amdgcn_update_dpp)
#define HAVE_DPP 1
#endif
#endif

#ifdef HAVE_ASYNC_LDS
// Address-space-qualified pointee types (attribute honored on typedefs).
typedef __attribute__((address_space(1))) v4i v4i_as1;  // global
typedef __attribute__((address_space(3))) v4i v4i_as3;  // LDS
#endif

__device__ inline unsigned short f2bf(float f) {
    union { float f; unsigned int u; } x; x.f = f;
    unsigned int r = x.u + 0x7FFFu + ((x.u >> 16) & 1u);  // round-to-nearest-even
    return (unsigned short)(r >> 16);
}

// Load a 16x32 bf16 A-fragment (or 32x16 B-fragment, same per-lane pattern):
// prow points at element (row_for_this_lane, k=0) with K contiguous.
__device__ inline v16bf load_frag(const unsigned short* prow, int h) {
    union { v16bf v; unsigned int u[8]; } f;
    const unsigned short* p = prow + h * 8;
#pragma unroll
    for (int i = 0; i < 4; ++i) f.u[i]     = *(const unsigned int*)(p + 2 * i);
#pragma unroll
    for (int i = 0; i < 4; ++i) f.u[4 + i] = *(const unsigned int*)(p + 16 + 2 * i);
    return f.v;
}

__device__ inline v8f wmma_bf16(v16bf a, v16bf b, v8f c) {
    return __builtin_amdgcn_wmma_f32_16x16x32_bf16(false, a, false, b, (short)0, c,
                                                   false, false);
}

// ----- 16-lane-row reductions: pure-VALU DPP butterfly (no LDS round trips) ---
#ifdef HAVE_DPP
template <int CTRL>
__device__ inline float dpp_shuf(float x) {
    int xi = __builtin_bit_cast(int, x);
    int r = __builtin_amdgcn_update_dpp(xi, xi, CTRL, 0xf, 0xf, false);
    return __builtin_bit_cast(float, r);
}
__device__ inline float redmax16(float x) {
    x = fmaxf(x, dpp_shuf<0xB1>(x));   // quad_perm(1,0,3,2)  xor 1
    x = fmaxf(x, dpp_shuf<0x4E>(x));   // quad_perm(2,3,0,1)  xor 2
    x = fmaxf(x, dpp_shuf<0x141>(x));  // row_half_mirror
    x = fmaxf(x, dpp_shuf<0x140>(x));  // row_mirror
    return x;
}
__device__ inline float redsum16(float x) {
    x += dpp_shuf<0xB1>(x);
    x += dpp_shuf<0x4E>(x);
    x += dpp_shuf<0x141>(x);
    x += dpp_shuf<0x140>(x);
    return x;
}
#else
__device__ inline float redmax16(float x) {
#pragma unroll
    for (int m = 1; m < 16; m <<= 1) x = fmaxf(x, __shfl_xor(x, m, 32));
    return x;
}
__device__ inline float redsum16(float x) {
#pragma unroll
    for (int m = 1; m < 16; m <<= 1) x += __shfl_xor(x, m, 32);
    return x;
}
#endif

// ---------------------------------------------------------------------------
// Kernel 1: sub + grouped 3x3 convs -> y1,y2 (bf16), layout [B][4096][512]
// ---------------------------------------------------------------------------
__global__ void __launch_bounds__(256, 1)
ca_dwconv_kernel(const float* __restrict__ x1, const float* __restrict__ x2,
                 const float* __restrict__ w1, const float* __restrict__ b1,
                 const float* __restrict__ w2, const float* __restrict__ b2,
                 unsigned short* __restrict__ y1, unsigned short* __restrict__ y2) {
    int t = blockIdx.x * blockDim.x + threadIdx.x;   // [0, B*4096*512)
    int o   = t & 511;
    int pix = (t >> 9) & (NN - 1);
    int b   = t >> 21;
    int hh = pix >> 6, ww = pix & 63;
    int g = o >> 1;
    bool isSub = (g < 128);
    int cbase = isSub ? (g * 2) : (g * 2 - 256);
    float acc1 = b1[o], acc2 = b2[o];
    const float* w1o = w1 + o * 18;
    const float* w2o = w2 + o * 18;
#pragma unroll
    for (int i = 0; i < 2; ++i) {
        int c = cbase + i;
        const float* p1 = x1 + ((size_t)b * CC + c) * NN;
        const float* p2 = x2 + ((size_t)b * CC + c) * NN;
#pragma unroll
        for (int kh = 0; kh < 3; ++kh) {
            int yy = hh + kh - 1;
            if ((unsigned)yy >= 64u) continue;
#pragma unroll
            for (int kw = 0; kw < 3; ++kw) {
                int xx = ww + kw - 1;
                if ((unsigned)xx >= 64u) continue;
                int idx = yy * 64 + xx;
                float a1 = p1[idx], a2 = p2[idx];
                float v1v, v2v;
                if (isSub) { float s = a1 - a2; v1v = s; v2v = s; }
                else       { v1v = a1; v2v = a2; }
                acc1 = fmaf(w1o[i * 9 + kh * 3 + kw], v1v, acc1);
                acc2 = fmaf(w2o[i * 9 + kh * 3 + kw], v2v, acc2);
            }
        }
    }
    y1[t] = f2bf(acc1);   // t == ((b*4096 + pix)*512 + o)
    y2[t] = f2bf(acc2);
}

// ---------------------------------------------------------------------------
// Kernel 2 (templated): WMMA GEMM  Out = epi( W[M][K](fp32) @ X[b][4096][K](bf16)^T )
// B tile async-copied K-contiguous into LDS (layout [col][K]).
// transpose_out=0: Out[b][M][4096]; =1: Out[b][4096][M] (both bf16).
// ---------------------------------------------------------------------------
template <int BM, int BN, int WM, int WN>
__global__ void __launch_bounds__(256, 1)
ca_gemm_wmma(const float* __restrict__ Wm, const float* __restrict__ bias,
             const unsigned short* __restrict__ X, unsigned short* __restrict__ Out,
             int M, int K, const float* __restrict__ bn_g,
             const float* __restrict__ bn_b, const float* __restrict__ bn_m,
             const float* __restrict__ bn_v, int transpose_out) {
    constexpr int BK = 32;
    constexpr int WAVES_M = BM / WM;
    constexpr int TM = WM / 16, TN = WN / 16;
    __shared__ __align__(16) unsigned short lA[BM][BK];  // [row][K]
    __shared__ __align__(16) unsigned short lB[BN][BK];  // [col][K]

    int tid = threadIdx.x;
    int lane = tid & 31, wave = tid >> 5;
    int h = lane >> 4, ln = lane & 15;
    int wm = wave % WAVES_M, wn = wave / WAVES_M;
    int n0 = blockIdx.x * BN;
    int m0 = blockIdx.y * BM;
    int bb = n0 >> 12;
    int nl0 = n0 & (NN - 1);
    const unsigned short* Xb = X + (size_t)bb * NN * K;

    v8f acc[TM][TN];
    v8f zc = {0.f, 0.f, 0.f, 0.f, 0.f, 0.f, 0.f, 0.f};
#pragma unroll
    for (int i = 0; i < TM; ++i)
#pragma unroll
        for (int j = 0; j < TN; ++j) acc[i][j] = zc;

    for (int k0 = 0; k0 < K; k0 += BK) {
        // A tile: fp32 weights -> bf16 LDS
        for (int e = tid; e < BM * BK; e += 256) {
            int r = e >> 5, c = e & 31;
            lA[r][c] = f2bf(Wm[(size_t)(m0 + r) * K + k0 + c]);
        }
        // B tile: K-contiguous 16-byte chunks, per-lane async copy to LDS
        for (int e = tid; e < BN * (BK / 8); e += 256) {
            int col = e >> 2, kp = e & 3;
            const unsigned short* src = Xb + (size_t)(nl0 + col) * K + k0 + kp * 8;
#ifdef HAVE_ASYNC_LDS
            __builtin_amdgcn_global_load_async_to_lds_b128(
                (v4i_as1*)(size_t)src,
                (v4i_as3*)(void*)&lB[col][kp * 8], 0, 0);
#else
            *(int4*)(&lB[col][kp * 8]) = *(const int4*)src;
#endif
        }
#ifdef HAVE_ASYNC_LDS
        asm volatile("s_wait_asynccnt 0x0" ::: "memory");
#endif
        __syncthreads();
        v16bf af[TM], bf[TN];
#pragma unroll
        for (int i = 0; i < TM; ++i)
            af[i] = load_frag(&lA[wm * WM + i * 16 + ln][0], h);
#pragma unroll
        for (int j = 0; j < TN; ++j)
            bf[j] = load_frag(&lB[wn * WN + j * 16 + ln][0], h);
#pragma unroll
        for (int i = 0; i < TM; ++i)
#pragma unroll
            for (int j = 0; j < TN; ++j) acc[i][j] = wmma_bf16(af[i], bf[j], acc[i][j]);
        __syncthreads();
    }

    // Epilogue. C-layout: lane ln = column, VGPR r -> row r + 8*h.
#pragma unroll
    for (int i = 0; i < TM; ++i) {
#pragma unroll
        for (int j = 0; j < TN; ++j) {
            int ncol = nl0 + wn * WN + j * 16 + ln;
#pragma unroll
            for (int r = 0; r < 8; ++r) {
                int m = m0 + wm * WM + i * 16 + r + 8 * h;
                float val = acc[i][j][r] + bias[m];
                if (bn_g != nullptr) {
                    float sc = bn_g[m] * rsqrtf(bn_v[m] + 1e-5f);
                    val = fmaxf(val * sc + (bn_b[m] - bn_m[m] * sc), 0.f);
                }
                size_t idx = transpose_out
                                 ? ((size_t)bb * NN * M + (size_t)ncol * M + m)
                                 : ((size_t)bb * M * NN + (size_t)m * NN + ncol);
                Out[idx] = f2bf(val);
            }
        }
    }
}

// ---------------------------------------------------------------------------
// Kernel 3: flash attention + fused  gamma*out + x1.
// qT,kT: [B][4096][32] bf16 (D contiguous)   V: [B][256][4096] bf16
// 8 waves/block: wave = (query-subtile qw in 0..3, channel-half cw in 0..1).
// ---------------------------------------------------------------------------
__global__ void __launch_bounds__(256, 1)
ca_flash_attn_kernel(const unsigned short* __restrict__ qT,
                     const unsigned short* __restrict__ kT,
                     const unsigned short* __restrict__ V,
                     const float* __restrict__ x1,
                     const float* __restrict__ gamma_p, float* __restrict__ out) {
    __shared__ unsigned short Pl[8][16][72];  // per-wave P tile, padded stride
    int tid = threadIdx.x;
    int lane = tid & 31, wave = tid >> 5;
    int h = lane >> 4, ln = lane & 15;
    int qw = wave & 3, cw = wave >> 2;
    int b = blockIdx.y;
    int q0 = blockIdx.x * 64 + qw * 16;
    int ch0 = cw * 128;

    const unsigned short* qTb = qT + (size_t)b * NN * DD;
    const unsigned short* kTb = kT + (size_t)b * NN * DD;
    const unsigned short* Vb  = V  + (size_t)b * CC * NN;

    v16bf qf = load_frag(qTb + (size_t)(q0 + ln) * DD, h);

    v8f zc = {0.f, 0.f, 0.f, 0.f, 0.f, 0.f, 0.f, 0.f};
    v8f acc[8];
#pragma unroll
    for (int ct = 0; ct < 8; ++ct) acc[ct] = zc;
    float mrow[8], lrow[8];
#pragma unroll
    for (int i = 0; i < 8; ++i) { mrow[i] = -1e30f; lrow[i] = 0.f; }

    for (int kt = 0; kt < NN; kt += 64) {
        if (kt + 64 < NN) {
            __builtin_prefetch(kTb + (size_t)(kt + 64 + ln) * DD, 0, 0);
        }
        // Preload all four k-fragments so loads overlap, then WMMA burst.
        v16bf kf[4];
#pragma unroll
        for (int j = 0; j < 4; ++j)
            kf[j] = load_frag(kTb + (size_t)(kt + j * 16 + ln) * DD, h);
        v8f S[4];
#pragma unroll
        for (int j = 0; j < 4; ++j) S[j] = wmma_bf16(qf, kf[j], zc);

        // Online softmax; P written straight to LDS (no P register array).
#pragma unroll
        for (int i = 0; i < 8; ++i) {
            float rm = fmaxf(fmaxf(S[0][i], S[1][i]), fmaxf(S[2][i], S[3][i]));
            rm = redmax16(rm);
            float mn = fmaxf(mrow[i], rm);
            float sc = __expf(mrow[i] - mn);
            mrow[i] = mn;
            float rs = 0.f;
#pragma unroll
            for (int j = 0; j < 4; ++j) {
                float p = __expf(S[j][i] - mn);
                Pl[wave][i + 8 * h][j * 16 + ln] = f2bf(p);
                rs += p;
            }
            rs = redsum16(rs);
            lrow[i] = lrow[i] * sc + rs;
#pragma unroll
            for (int ct = 0; ct < 8; ++ct) acc[ct][i] *= sc;
        }
        // out += P @ V  (two 32-key chunks x 8 channel tiles; wave-local LDS)
#pragma unroll
        for (int c = 0; c < 2; ++c) {
            v16bf pf = load_frag(&Pl[wave][ln][c * 32], h);
            v16bf vf[4];
#pragma unroll
            for (int ct = 0; ct < 8; ct += 4) {
#pragma unroll
                for (int u = 0; u < 4; ++u)
                    vf[u] = load_frag(
                        Vb + (size_t)(ch0 + (ct + u) * 16 + ln) * NN + kt + c * 32,
                        h);
#pragma unroll
                for (int u = 0; u < 4; ++u)
                    acc[ct + u] = wmma_bf16(pf, vf[u], acc[ct + u]);
            }
        }
    }

    float g = gamma_p[0];
    float rl[8];
#pragma unroll
    for (int i = 0; i < 8; ++i) rl[i] = 1.f / lrow[i];
#pragma unroll
    for (int ct = 0; ct < 8; ++ct) {
        int ch = ch0 + ct * 16 + ln;
        const float* xrow = x1 + ((size_t)b * CC + ch) * NN;
        float* orow = out + ((size_t)b * CC + ch) * NN;
#pragma unroll
        for (int i = 0; i < 8; ++i) {
            int qq = q0 + i + 8 * h;
            orow[qq] = g * acc[ct][i] * rl[i] + xrow[qq];
        }
    }
}

// ---------------------------------------------------------------------------
// Host launcher
// ---------------------------------------------------------------------------
extern "C" void kernel_launch(void* const* d_in, const int* in_sizes, int n_in,
                              void* d_out, int out_size, void* d_ws, size_t ws_size,
                              hipStream_t stream) {
    const float* x1    = (const float*)d_in[0];
    const float* x2    = (const float*)d_in[1];
    const float* w1_dw = (const float*)d_in[2];
    const float* b1_dw = (const float*)d_in[3];
    const float* w1_pw = (const float*)d_in[4];
    const float* b1_pw = (const float*)d_in[5];
    const float* bn1_g = (const float*)d_in[6];
    const float* bn1_b = (const float*)d_in[7];
    const float* bn1_m = (const float*)d_in[8];
    const float* bn1_v = (const float*)d_in[9];
    const float* w2_dw = (const float*)d_in[10];
    const float* b2_dw = (const float*)d_in[11];
    const float* w2_pw = (const float*)d_in[12];
    const float* b2_pw = (const float*)d_in[13];
    const float* bn2_g = (const float*)d_in[14];
    const float* bn2_b = (const float*)d_in[15];
    const float* bn2_m = (const float*)d_in[16];
    const float* bn2_v = (const float*)d_in[17];
    const float* wq    = (const float*)d_in[18];
    const float* bq    = (const float*)d_in[19];
    const float* wk    = (const float*)d_in[20];
    const float* bk    = (const float*)d_in[21];
    const float* wv    = (const float*)d_in[22];
    const float* bv    = (const float*)d_in[23];
    const float* gamma = (const float*)d_in[24];

    // Workspace layout (bf16 = unsigned short), all regions 16B-aligned
    unsigned short* y1  = (unsigned short*)d_ws;                 // [B][4096][512]
    unsigned short* y2  = y1 + (size_t)BB * NN * 512;
    unsigned short* x3  = y2 + (size_t)BB * NN * 512;            // [B][4096][256]
    unsigned short* x4  = x3 + (size_t)BB * NN * CC;
    unsigned short* qT  = x4 + (size_t)BB * NN * CC;             // [B][4096][32]
    unsigned short* kTp = qT + (size_t)BB * NN * DD;
    unsigned short* Vp  = kTp + (size_t)BB * NN * DD;            // [B][256][4096]

    // 1) fused sub + grouped 3x3 convs
    {
        int total = BB * 512 * NN;
        ca_dwconv_kernel<<<total / 256, 256, 0, stream>>>(x1, x2, w1_dw, b1_dw,
                                                          w2_dw, b2_dw, y1, y2);
    }
    // 2) pointwise convs + BN + ReLU -> x3, x4 in [b][n][256]  (M=256,K=512)
    {
        dim3 g2(16384 / 128, 256 / 128);
        ca_gemm_wmma<128, 128, 64, 32><<<g2, 256, 0, stream>>>(
            w1_pw, b1_pw, y1, x3, 256, 512, bn1_g, bn1_b, bn1_m, bn1_v, 1);
        ca_gemm_wmma<128, 128, 64, 32><<<g2, 256, 0, stream>>>(
            w2_pw, b2_pw, y2, x4, 256, 512, bn2_g, bn2_b, bn2_m, bn2_v, 1);
    }
    // 3) v = Wv@x3 -> [b][256][n];  q = Wq@x4, k = Wk@x3 -> [b][n][32]
    {
        dim3 gv(16384 / 128, 256 / 128);
        ca_gemm_wmma<128, 128, 64, 32><<<gv, 256, 0, stream>>>(
            wv, bv, x3, Vp, 256, 256, nullptr, nullptr, nullptr, nullptr, 0);
        dim3 gq(16384 / 128, 1);
        ca_gemm_wmma<32, 128, 32, 16><<<gq, 256, 0, stream>>>(
            wq, bq, x4, qT, 32, 256, nullptr, nullptr, nullptr, nullptr, 1);
        ca_gemm_wmma<32, 128, 32, 16><<<gq, 256, 0, stream>>>(
            wk, bk, x3, kTp, 32, 256, nullptr, nullptr, nullptr, nullptr, 1);
    }
    // 4) flash attention + gamma*out + x1
    {
        dim3 ga(NN / 64, BB);
        ca_flash_attn_kernel<<<ga, 256, 0, stream>>>(qT, kTp, Vp, x1, gamma,
                                                     (float*)d_out);
    }
}
